// Attention_33681133535344
// MI455X (gfx1250) — matrix operand order repaired and checked by reading, not compiled, not run on hardware
//
#include <hip/hip_runtime.h>
#include <hip/hip_bf16.h>
#include <math.h>

// ---------------------------------------------------------------------------
// Attention QKV projection + RMSNorm + RoPE + head transpose/repeat (MI455X)
//
// B=2, T=4096, D=4096, H=32, HKV=8, DH=128
//   q = rope(rmsnorm(x@Wq, gq))  -> (B,H,T,DH)
//   k = rope(rmsnorm(x@Wk, gk))  -> repeat to (B,H,T,DH)
//   v = x@Wv                     -> repeat to (B,H,T,DH)
//
// bf16 WMMA GEMM (v_wmma_f32_16x16x32_bf16), async global->LDS tile staging
// (global_load_async_to_lds_b128 + s_wait_asynccnt), double-buffered LDS,
// pre-transposed bf16 weights so both tiles stage with coalesced 16B copies.
// ---------------------------------------------------------------------------

#define B_    2
#define T_    4096
#define D_    4096
#define H_    32
#define HKV_  8
#define DH_   128
#define M_    (B_ * T_)        // 8192 rows
#define NKV_  (HKV_ * DH_)     // 1024
#define EPS_  1e-5f
#define LN_ROPE_BASE 9.210340371976184f   // ln(10000)

#define BK_   64               // K-step per LDS buffer

typedef __attribute__((ext_vector_type(16))) __bf16 v16bf;
typedef __attribute__((ext_vector_type(8)))  float  v8f;

__device__ __forceinline__ __bf16 f2bf(float f) {
    unsigned u = __builtin_bit_cast(unsigned, f);
    unsigned r = (u + 0x7FFFu + ((u >> 16) & 1u)) >> 16;
    unsigned short s = (unsigned short)r;
    return __builtin_bit_cast(__bf16, s);
}

// ------------------------------ fp32 -> bf16 (RNE), x4 ---------------------
__global__ __launch_bounds__(256)
void cast_f32_to_bf16_v4(const float* __restrict__ src,
                         __bf16* __restrict__ dst, long long n4) {
    long long i      = (long long)blockIdx.x * blockDim.x + threadIdx.x;
    long long stride = (long long)gridDim.x * blockDim.x;
    union P { __bf16 e[4]; uint2 u; };
    for (; i < n4; i += stride) {
        float4 f = ((const float4*)src)[i];
        P p;
        p.e[0] = f2bf(f.x); p.e[1] = f2bf(f.y);
        p.e[2] = f2bf(f.z); p.e[3] = f2bf(f.w);
        ((uint2*)dst)[i] = p.u;
    }
}

// ---------------- weights: fp32 (KxN) -> bf16 transposed (NxK) -------------
__global__ __launch_bounds__(256)
void transpose_cast_w(const float* __restrict__ W, __bf16* __restrict__ Wt,
                      int K, int N) {
    __shared__ __bf16 tile[32][33];          // +1 pad: no bank conflicts
    const int tx = threadIdx.x & 31;
    const int ty = threadIdx.x >> 5;         // 0..7
    const int k0 = blockIdx.y * 32;
    const int n0 = blockIdx.x * 32;
#pragma unroll
    for (int r = 0; r < 4; ++r) {
        const int kr = ty + r * 8;
        tile[kr][tx] = f2bf(W[(size_t)(k0 + kr) * N + n0 + tx]);
    }
    __syncthreads();
#pragma unroll
    for (int r = 0; r < 4; ++r) {
        const int nr = ty + r * 8;
        Wt[(size_t)(n0 + nr) * K + k0 + tx] = tile[tx][nr];
    }
}

// ------------------- async global -> LDS tile staging ----------------------
// Stages a 128 x BK_ bf16 tile (16 KB) with 4 x global_load_async_to_lds_b128
// per thread (ASYNCcnt-tracked; data never touches VGPRs).
__device__ __forceinline__ void stage_tile_async(const __bf16* __restrict__ g,
                                                 int ldg, unsigned lds_base,
                                                 int tid) {
#pragma unroll
    for (int c = 0; c < 4; ++c) {
        const int ch  = tid + c * 256;        // 0..1023 chunks of 8 bf16
        const int row = ch >> 3;              // 128 rows
        const int kc  = (ch & 7) * 8;         // 64 cols / 8
        const unsigned long long gp =
            (unsigned long long)(uintptr_t)(g + (size_t)row * ldg + kc);
        const unsigned lp = lds_base + (unsigned)(row * BK_ + kc) * 2u;
        asm volatile("global_load_async_to_lds_b128 %0, %1, off"
                     :: "v"(lp), "v"(gp)
                     : "memory");
    }
}

// ------------------------------ WMMA GEMM ----------------------------------
// C(MxN) = A(MxK) * Wt(NxK)^T, bf16 in, fp32 out.
// Workgroup tile 128x128, K-step 64, double-buffered async LDS staging.
// 8 wave32s: 4 (M) x 2 (N); each wave owns 32x64 = 2x4 WMMA tiles.
// MODE 0: write fp32 row-major into C.
// MODE 1: V path — write directly to out in (B,H,T,DH) layout, 4x KV repeat.
template <int MODE>
__global__ __launch_bounds__(256)
void gemm_bf16_wmma(const __bf16* __restrict__ A, const __bf16* __restrict__ Wt,
                    float* __restrict__ C, int N, int K) {
    __shared__ __bf16 ldsA[2][128 * BK_];    // [m][k] row-major, 16 KB each
    __shared__ __bf16 ldsB[2][128 * BK_];    // [n][k] row-major, 16 KB each

    const int tid   = threadIdx.x;
    const int lane  = tid & 31;
    const int wid   = tid >> 5;
    const int wm    = wid & 3;     // wave M index (0..3)
    const int wn    = wid >> 2;    // wave N index (0..1)
    const int m0    = blockIdx.y * 128;
    const int n0    = blockIdx.x * 128;

    const int frow  = lane & 15;   // row (A) / col (B,D) within 16x16 tile
    const int fhalf = lane >> 4;   // lane half selects K-subrange / M-half
    const int a_kb  = fhalf * 8;   // A frag: v0-3 <- K[a_kb..a_kb+7]
    const int b_kb  = fhalf * 8;   // B frag: v0-3 <- K[b_kb..b_kb+7]

    const unsigned baseA0 = (unsigned)(uintptr_t)&ldsA[0][0];
    const unsigned baseA1 = (unsigned)(uintptr_t)&ldsA[1][0];
    const unsigned baseB0 = (unsigned)(uintptr_t)&ldsB[0][0];
    const unsigned baseB1 = (unsigned)(uintptr_t)&ldsB[1][0];

    const v8f vzero = {0.f, 0.f, 0.f, 0.f, 0.f, 0.f, 0.f, 0.f};
    v8f acc[2][4];
#pragma unroll
    for (int i = 0; i < 2; ++i)
#pragma unroll
        for (int j = 0; j < 4; ++j) acc[i][j] = vzero;

    const __bf16* gA = A  + (size_t)m0 * K;
    const __bf16* gB = Wt + (size_t)n0 * K;

    // prologue: stage K-tile 0 into buffer 0
    stage_tile_async(gA, K, baseA0, tid);
    stage_tile_async(gB, K, baseB0, tid);

    const int KT = K / BK_;
    for (int kt = 0; kt < KT; ++kt) {
        const int buf = kt & 1;

        // publish async LDS writes, then cross-wave barrier
        asm volatile("s_wait_asynccnt 0x0" ::: "memory");
        __syncthreads();

        // stage next K-tile into the other buffer (overlaps with WMMA below)
        if (kt + 1 < KT) {
            const size_t koff = (size_t)(kt + 1) * BK_;
            stage_tile_async(gA + koff, K, buf ? baseA0 : baseA1, tid);
            stage_tile_async(gB + koff, K, buf ? baseB0 : baseB1, tid);
        }

        const __bf16* pa = &ldsA[buf][0];
        const __bf16* pb = &ldsB[buf][0];
#pragma unroll
        for (int ks = 0; ks < BK_; ks += 32) {
            union AF { v16bf v; uint4 u[2]; } af[2];
            union BF { v16bf v; uint4 u[2]; } bf[4];
#pragma unroll
            for (int mi = 0; mi < 2; ++mi) {
                const __bf16* p = pa + (wm * 32 + mi * 16 + frow) * BK_ + ks;
                af[mi].u[0] = *(const uint4*)(p + a_kb);
                af[mi].u[1] = *(const uint4*)(p + a_kb + 16);
            }
#pragma unroll
            for (int ni = 0; ni < 4; ++ni) {
                const __bf16* p = pb + (wn * 64 + ni * 16 + frow) * BK_ + ks;
                bf[ni].u[0] = *(const uint4*)(p + b_kb);
                bf[ni].u[1] = *(const uint4*)(p + b_kb + 16);
            }
#pragma unroll
            for (int mi = 0; mi < 2; ++mi)
#pragma unroll
                for (int ni = 0; ni < 4; ++ni)
                    acc[mi][ni] = __builtin_amdgcn_wmma_f32_16x16x32_bf16(
                        false, af[mi].v, false, bf[ni].v,
                        (short)0, acc[mi][ni], false, false);
        }
        // drain reads of `buf` before it is re-staged at iteration kt+1
        __syncthreads();
    }

    // --- epilogue: D layout = vgpr r -> M = r + 8*fhalf, N = frow ---
#pragma unroll
    for (int mi = 0; mi < 2; ++mi) {
#pragma unroll
        for (int ni = 0; ni < 4; ++ni) {
            const int n     = n0 + wn * 64 + ni * 16 + frow;
            const int mbase = m0 + wm * 32 + mi * 16 + fhalf * 8;
            if (MODE == 0) {
#pragma unroll
                for (int r = 0; r < 8; ++r)
                    C[(size_t)(mbase + r) * N + n] = acc[mi][ni][r];
            } else {
                const int hkv = n >> 7;       // n / DH
                const int d   = n & (DH_ - 1);
#pragma unroll
                for (int r = 0; r < 8; ++r) {
                    const int m = mbase + r;
                    const int b = m >> 12;    // m / T
                    const int t = m & (T_ - 1);
                    const float val = acc[mi][ni][r];
#pragma unroll
                    for (int rep = 0; rep < 4; ++rep) {
                        const int h = hkv * 4 + rep;
                        C[(((size_t)b * H_ + h) * T_ + t) * DH_ + d] = val;
                    }
                }
            }
        }
    }
}

// ------------------------- per-row inverse RMS -----------------------------
__global__ __launch_bounds__(256)
void row_inv_rms(const float* __restrict__ Y, float* __restrict__ out,
                 int width) {
    __shared__ float red[256];
    const int row = blockIdx.x;
    const float* p = Y + (size_t)row * width;
    float s = 0.f;
    for (int i = threadIdx.x; i < width; i += 256) {
        float v = p[i];
        s += v * v;
    }
    red[threadIdx.x] = s;
    __syncthreads();
    for (int off = 128; off > 0; off >>= 1) {
        if (threadIdx.x < off) red[threadIdx.x] += red[threadIdx.x + off];
        __syncthreads();
    }
    if (threadIdx.x == 0) out[row] = rsqrtf(red[0] / (float)width + EPS_);
}

// -------------------- Q: norm + rope + transpose ---------------------------
__global__ __launch_bounds__(256)
void rope_write_q(const float* __restrict__ Ylin, const float* __restrict__ inv,
                  const float* __restrict__ g, float* __restrict__ out) {
    const long long total = (long long)M_ * H_ * (DH_ / 2);
    long long idx = (long long)blockIdx.x * blockDim.x + threadIdx.x;
    if (idx >= total) return;
    const int i   = (int)(idx & 63);
    const int h   = (int)((idx >> 6) & (H_ - 1));
    const int row = (int)(idx >> 11);           // 64*H = 2048 = 2^11
    const int t   = row & (T_ - 1);
    const int b   = row >> 12;
    const int c0  = h * DH_ + 2 * i;

    const float s  = inv[row];
    const float y1 = Ylin[(size_t)row * D_ + c0]     * s * g[c0];
    const float y2 = Ylin[(size_t)row * D_ + c0 + 1] * s * g[c0 + 1];

    const float freq = __expf(-(float)(2 * i) * (LN_ROPE_BASE / (float)DH_));
    float sn, cs;
    __sincosf((float)t * freq, &sn, &cs);
    const size_t ob = (((size_t)b * H_ + h) * T_ + t) * DH_ + 2 * i;
    out[ob]     = y1 * cs - y2 * sn;
    out[ob + 1] = y1 * sn + y2 * cs;
}

// -------------------- K: norm + rope + transpose + 4x repeat ---------------
__global__ __launch_bounds__(256)
void rope_write_k(const float* __restrict__ Ylin, const float* __restrict__ inv,
                  const float* __restrict__ g, float* __restrict__ out) {
    const long long total = (long long)M_ * HKV_ * (DH_ / 2);
    long long idx = (long long)blockIdx.x * blockDim.x + threadIdx.x;
    if (idx >= total) return;
    const int i   = (int)(idx & 63);
    const int hk  = (int)((idx >> 6) & (HKV_ - 1));
    const int row = (int)(idx >> 9);            // 64*HKV = 512 = 2^9
    const int t   = row & (T_ - 1);
    const int b   = row >> 12;
    const int c0  = hk * DH_ + 2 * i;

    const float s  = inv[row];
    const float y1 = Ylin[(size_t)row * NKV_ + c0]     * s * g[c0];
    const float y2 = Ylin[(size_t)row * NKV_ + c0 + 1] * s * g[c0 + 1];

    const float freq = __expf(-(float)(2 * i) * (LN_ROPE_BASE / (float)DH_));
    float sn, cs;
    __sincosf((float)t * freq, &sn, &cs);
    const float r1 = y1 * cs - y2 * sn;
    const float r2 = y1 * sn + y2 * cs;
#pragma unroll
    for (int rep = 0; rep < 4; ++rep) {
        const int h = hk * 4 + rep;
        const size_t ob = (((size_t)b * H_ + h) * T_ + t) * DH_ + 2 * i;
        out[ob]     = r1;
        out[ob + 1] = r2;
    }
}

// ---------------------------------------------------------------------------
extern "C" void kernel_launch(void* const* d_in, const int* in_sizes, int n_in,
                              void* d_out, int out_size, void* d_ws,
                              size_t ws_size, hipStream_t stream) {
    (void)in_sizes; (void)n_in; (void)out_size; (void)ws_size;

    const float* x  = (const float*)d_in[0];
    const float* Wq = (const float*)d_in[1];
    const float* Wk = (const float*)d_in[2];
    const float* Wv = (const float*)d_in[3];
    const float* gq = (const float*)d_in[4];
    const float* gk = (const float*)d_in[5];
    float* out = (float*)d_out;

    const size_t QSZ = (size_t)B_ * H_ * T_ * DH_;   // 33,554,432 elems

    // workspace carve-out (~280 MB total)
    char*  ws  = (char*)d_ws;
    size_t off = 0;
    auto carve = [&](size_t bytes) -> void* {
        void* p = ws + off;
        off = (off + bytes + 255) & ~(size_t)255;
        return p;
    };
    __bf16* xb   = (__bf16*)carve((size_t)M_ * D_   * sizeof(__bf16));
    __bf16* wqt  = (__bf16*)carve((size_t)D_ * D_   * sizeof(__bf16));
    __bf16* wkt  = (__bf16*)carve((size_t)D_ * NKV_ * sizeof(__bf16));
    __bf16* wvt  = (__bf16*)carve((size_t)D_ * NKV_ * sizeof(__bf16));
    float*  qlin = (float*) carve((size_t)M_ * D_   * sizeof(float));
    float*  klin = (float*) carve((size_t)M_ * NKV_ * sizeof(float));
    float*  invq = (float*) carve((size_t)M_ * sizeof(float));
    float*  invk = (float*) carve((size_t)M_ * sizeof(float));

    // 1) activations -> bf16 (vectorized); weights -> bf16 transposed (NxK)
    cast_f32_to_bf16_v4<<<2048, 256, 0, stream>>>(x, xb,
                                                  (long long)M_ * D_ / 4);
    transpose_cast_w<<<dim3(D_ / 32,   D_ / 32), 256, 0, stream>>>(Wq, wqt,
                                                                   D_, D_);
    transpose_cast_w<<<dim3(NKV_ / 32, D_ / 32), 256, 0, stream>>>(Wk, wkt,
                                                                   D_, NKV_);
    transpose_cast_w<<<dim3(NKV_ / 32, D_ / 32), 256, 0, stream>>>(Wv, wvt,
                                                                   D_, NKV_);

    // 2) WMMA GEMMs (grid: N/128 x M/128)
    gemm_bf16_wmma<0><<<dim3(D_ / 128,   M_ / 128), 256, 0, stream>>>(
        xb, wqt, qlin, D_, D_);
    gemm_bf16_wmma<0><<<dim3(NKV_ / 128, M_ / 128), 256, 0, stream>>>(
        xb, wkt, klin, NKV_, D_);
    gemm_bf16_wmma<1><<<dim3(NKV_ / 128, M_ / 128), 256, 0, stream>>>(
        xb, wvt, out + 2 * QSZ, NKV_, D_);   // V straight to output (repeated)

    // 3) RMS statistics
    row_inv_rms<<<M_, 256, 0, stream>>>(qlin, invq, D_);
    row_inv_rms<<<M_, 256, 0, stream>>>(klin, invk, NKV_);

    // 4) norm + rope + transpose (+repeat for K)
    {
        const long long tq = (long long)M_ * H_ * (DH_ / 2);
        rope_write_q<<<(unsigned)((tq + 255) / 256), 256, 0, stream>>>(
            qlin, invq, gq, out);
        const long long tk = (long long)M_ * HKV_ * (DH_ / 2);
        rope_write_k<<<(unsigned)((tk + 255) / 256), 256, 0, stream>>>(
            klin, invk, gk, out + QSZ);
    }
}
